// GraphConvolutionalLayer_78073915507128
// MI455X (gfx1250) — matrix operand order repaired and checked by reading, not compiled, not run on hardware
//
#include <hip/hip_runtime.h>

#define DEV __device__ __forceinline__

// ---------------- problem constants ----------------
#define NN 1024
#define DD 128
#define BN_EPS 1e-5f

typedef __attribute__((ext_vector_type(16))) __bf16 v16bf;
typedef __attribute__((ext_vector_type(8)))  __bf16 v8bf;
typedef __attribute__((ext_vector_type(4)))  __bf16 v4bf;
typedef __attribute__((ext_vector_type(8)))  float  v8f;

#if __has_builtin(__builtin_amdgcn_tanhf)
#define fast_tanh(x) __builtin_amdgcn_tanhf(x)
#else
#define fast_tanh(x) tanhf(x)
#endif

// round-to-nearest-even f32 -> bf16
DEV __bf16 f2bf(float f) {
  unsigned int u = __builtin_bit_cast(unsigned int, f);
  u += 0x7FFFu + ((u >> 16) & 1u);
  unsigned short h = (unsigned short)(u >> 16);
  return __builtin_bit_cast(__bf16, h);
}

// A operand: 16x32 (MxK) bf16 tile from row-major bf16 matrix P (leading dim ld)
// lane L (0-15): M=L, kbase=0 ; lane L+16: M=L, kbase=8
// element e -> K = kbase + (e&7) + (e>=8 ? 16 : 0)  => two contiguous 16B chunks
DEV v16bf load_a_tile(const __bf16* __restrict__ P, int ld, int row0, int k0, int lane) {
  int m  = lane & 15;
  int kb = (lane >> 4) << 3;
  const __bf16* p = P + (size_t)(row0 + m) * ld + k0 + kb;
  v8bf lo = *(const v8bf*)(p);       // K = kb .. kb+7
  v8bf hi = *(const v8bf*)(p + 16);  // K = kb+16 .. kb+23
  v16bf a;
#pragma unroll
  for (int e = 0; e < 8; ++e) { a[e] = lo[e]; a[8 + e] = hi[e]; }
  return a;
}

// B operand: 32x16 (KxN) bf16 tile: lane L -> col n0+(L&15), K order same as A tile
DEV v16bf load_b_tile(const __bf16* __restrict__ P, int ld, int k0, int n0, int lane) {
  int kb = (lane >> 4) << 3;
  const __bf16* p = P + (size_t)(k0 + kb) * ld + n0 + (lane & 15);
  v16bf b;
#pragma unroll
  for (int e = 0; e < 8; ++e) { b[e] = p[(size_t)e * ld]; b[8 + e] = p[(size_t)(16 + e) * ld]; }
  return b;
}

DEV v8f wmma_bf16(v16bf a, v16bf b, v8f c) {
  return __builtin_amdgcn_wmma_f32_16x16x32_bf16(false, a, false, b, (short)0, c, false, false);
}

// ---------------- K-1: bulk f32 -> bf16 conversion (vectorized x4) ------
__global__ void k_cvt4(const float* __restrict__ in, __bf16* __restrict__ out, int n4) {
  int i = blockIdx.x * 256 + threadIdx.x;
  if (i >= n4) return;
  float4 v = ((const float4*)in)[i];
  v4bf o;
  o[0] = f2bf(v.x); o[1] = f2bf(v.y); o[2] = f2bf(v.z); o[3] = f2bf(v.w);
  ((v4bf*)out)[i] = o;
}

// ---------------- K0: compose attention weights into W ------------------
// Wsrc = W_comm @ Wa1[:DD], Wtgt = W_comm @ Wa1[DD:]  (f32 math, bf16 store)
__global__ void k_compose(const float* __restrict__ Wcomm,
                          const float* __restrict__ Wa1,
                          __bf16* __restrict__ Wsrcb, __bf16* __restrict__ Wtgtb) {
  int idx  = blockIdx.x * blockDim.x + threadIdx.x;   // 0 .. 32767
  int half = idx >> 14;                                // 0 = src, 1 = tgt
  int rem  = idx & 16383;
  int a = rem >> 7, c = rem & 127;
  float acc = 0.f;
#pragma unroll 8
  for (int b = 0; b < DD; ++b)
    acc += Wcomm[a * DD + b] * Wa1[(half * DD + b) * DD + c];
  (half ? Wtgtb : Wsrcb)[a * DD + c] = f2bf(acc);
}

// ---------------- K1: five GEMMs [1024,128] x [128,128] via WMMA --------
// c=0: S0=X@Wself+bias (f32) ; c=1: T=X@Wcomm (bf16) ; c=2: E=X@Wexec (bf16)
// c=3: SRC=X@Wsrc+ba1 (f32)  ; c=4: TGT=X@Wtgt (f32)
__global__ __launch_bounds__(256)
void k_gemm_x5(const __bf16* __restrict__ Xb,
               const __bf16* __restrict__ Wselfb, const __bf16* __restrict__ Wcommb,
               const __bf16* __restrict__ Wexecb, const __bf16* __restrict__ Wsrcb,
               const __bf16* __restrict__ Wtgtb,
               const float* __restrict__ bias, const float* __restrict__ ba1,
               float* __restrict__ S0, __bf16* __restrict__ Tb, __bf16* __restrict__ Eb,
               float* __restrict__ SRC, float* __restrict__ TGT) {
  int wid  = threadIdx.x >> 5;
  int lane = threadIdx.x & 31;
  int gw   = blockIdx.x * 8 + wid;     // 0 .. 2559
  int c    = gw >> 9;                  // which output matrix
  int tile = gw & 511;
  int mt = tile >> 3, nt = tile & 7;

  const __bf16* W; float* Df = nullptr; __bf16* Db = nullptr; const float* g = nullptr;
  switch (c) {
    case 0:  W = Wselfb; Df = S0;  g = bias; break;
    case 1:  W = Wcommb; Db = Tb;  break;
    case 2:  W = Wexecb; Db = Eb;  break;
    case 3:  W = Wsrcb;  Df = SRC; g = ba1;  break;
    default: W = Wtgtb;  Df = TGT; break;
  }

  v8f acc = {};
#pragma unroll
  for (int k0 = 0; k0 < DD; k0 += 32) {
    v16bf a = load_a_tile(Xb, DD, mt * 16, k0, lane);
    v16bf b = load_b_tile(W, DD, k0, nt * 16, lane);
    acc = wmma_bf16(a, b, acc);
  }

  int col = nt * 16 + (lane & 15);
  if (Db) {  // wave-uniform branch
#pragma unroll
    for (int r = 0; r < 8; ++r) {
      int row = mt * 16 + r + ((lane >> 4) << 3);
      Db[row * DD + col] = f2bf(acc[r]);
    }
  } else {
    float add = g ? g[col] : 0.0f;
#pragma unroll
    for (int r = 0; r < 8; ++r) {
      int row = mt * 16 + r + ((lane >> 4) << 3);
      Df[row * DD + col] = acc[r] + add;
    }
  }
}

// ---------------- K2: pairwise tanh-MLP attention + row softmax ---------
// One block per row i. score_ij = sigmoid(sum_d tanh(SRC[i,d]+TGT[j,d])*Wa2[d] + ba2)
// (ba1 folded into SRC). att = softmax_j(score * adj_comm), stored bf16.
__global__ __launch_bounds__(256)
void k_att(const float* __restrict__ SRC, const float* __restrict__ TGT,
           const float* __restrict__ adj, const float* __restrict__ Wa2,
           const float* __restrict__ ba2, __bf16* __restrict__ ATTb) {
  __shared__ float sSrc[DD];
  __shared__ float sW[DD];
  __shared__ float sc[NN];
  __shared__ float red[256];

  int i = blockIdx.x, t = threadIdx.x;
  if (t < DD) { sSrc[t] = SRC[i * DD + t]; sW[t] = Wa2[t]; }
  __syncthreads();
  float b2 = ba2[0];

  for (int jb = 0; jb < 4; ++jb) {
    int j = jb * 256 + t;
    const float4* tg = (const float4*)(TGT + (size_t)j * DD);
    float acc = 0.f;
#pragma unroll 8
    for (int d4 = 0; d4 < DD / 4; ++d4) {
      float4 v = tg[d4];
      int d = d4 * 4;
      acc += fast_tanh(sSrc[d + 0] + v.x) * sW[d + 0];
      acc += fast_tanh(sSrc[d + 1] + v.y) * sW[d + 1];
      acc += fast_tanh(sSrc[d + 2] + v.z) * sW[d + 2];
      acc += fast_tanh(sSrc[d + 3] + v.w) * sW[d + 3];
    }
    float s = 1.0f / (1.0f + __expf(-(acc + b2)));
    sc[j] = s * adj[(size_t)i * NN + j];
  }
  __syncthreads();

  // softmax over j (1024 values, 4 per thread)
  float m = fmaxf(fmaxf(sc[t], sc[t + 256]), fmaxf(sc[t + 512], sc[t + 768]));
  red[t] = m; __syncthreads();
  for (int s = 128; s > 0; s >>= 1) {
    if (t < s) red[t] = fmaxf(red[t], red[t + s]);
    __syncthreads();
  }
  float mx = red[0];
  __syncthreads();

  float lsum = 0.f;
  for (int jb = 0; jb < 4; ++jb) {
    int j = jb * 256 + t;
    float e = __expf(sc[j] - mx);
    sc[j] = e; lsum += e;
  }
  red[t] = lsum; __syncthreads();
  for (int s = 128; s > 0; s >>= 1) {
    if (t < s) red[t] += red[t + s];
    __syncthreads();
  }
  float inv = 1.0f / red[0];
  for (int jb = 0; jb < 4; ++jb) {
    int j = jb * 256 + t;
    ATTb[(size_t)i * NN + j] = f2bf(sc[j] * inv);
  }
}

// ---------------- K3: PRE = S0 + ATT@T + ADJ_EXEC@E via WMMA ------------
// Each wave: one 16-row m-tile x four 16-col n-tiles (64 cols), K = 1024.
__global__ __launch_bounds__(256)
void k_out_gemm(const __bf16* __restrict__ ATTb, const __bf16* __restrict__ Tb,
                const __bf16* __restrict__ ADJb, const __bf16* __restrict__ Eb,
                const float* __restrict__ S0, float* __restrict__ PRE) {
  int wid  = threadIdx.x >> 5;
  int lane = threadIdx.x & 31;
  int gw   = blockIdx.x * 8 + wid;   // 0 .. 127
  int mt = gw >> 1;                  // 0 .. 63
  int nh = (gw & 1) * 64;            // column half: 0 or 64

  v8f acc[4] = {};
  for (int k0 = 0; k0 < NN; k0 += 32) {
    v16bf aA = load_a_tile(ATTb, NN, mt * 16, k0, lane);
    v16bf aE = load_a_tile(ADJb, NN, mt * 16, k0, lane);
#pragma unroll
    for (int n = 0; n < 4; ++n) {
      v16bf bT = load_b_tile(Tb, DD, k0, nh + n * 16, lane);
      acc[n] = wmma_bf16(aA, bT, acc[n]);
      v16bf bE = load_b_tile(Eb, DD, k0, nh + n * 16, lane);
      acc[n] = wmma_bf16(aE, bE, acc[n]);
    }
  }

#pragma unroll
  for (int n = 0; n < 4; ++n) {
    int col = nh + n * 16 + (lane & 15);
#pragma unroll
    for (int r = 0; r < 8; ++r) {
      int row = mt * 16 + r + ((lane >> 4) << 3);
      PRE[row * DD + col] = acc[n][r] + S0[row * DD + col];
    }
  }
}

// ---------------- K4: BatchNorm training statistics ---------------------
__global__ void k_stats(const float* __restrict__ PRE,
                        float* __restrict__ MEAN, float* __restrict__ RSTD) {
  int d = threadIdx.x;  // 128 threads, one per channel
  float s = 0.f, ss = 0.f;
  for (int i = 0; i < NN; ++i) {
    float x = PRE[i * DD + d];
    s += x; ss += x * x;
  }
  float mean = s * (1.0f / NN);
  float var  = ss * (1.0f / NN) - mean * mean;
  MEAN[d] = mean;
  RSTD[d] = rsqrtf(var + BN_EPS);
}

// ---------------- K5: normalize + affine + ReLU -------------------------
__global__ void k_bn_relu(const float* __restrict__ PRE,
                          const float* __restrict__ MEAN, const float* __restrict__ RSTD,
                          const float* __restrict__ gamma, const float* __restrict__ beta,
                          float* __restrict__ OUT) {
  int idx = blockIdx.x * 256 + threadIdx.x;
  int d = idx & (DD - 1);
  float y = (PRE[idx] - MEAN[d]) * RSTD[d] * gamma[d] + beta[d];
  OUT[idx] = fmaxf(y, 0.0f);
}

// ---------------- launch -------------------------------------------------
extern "C" void kernel_launch(void* const* d_in, const int* in_sizes, int n_in,
                              void* d_out, int out_size, void* d_ws, size_t ws_size,
                              hipStream_t stream) {
  const float* X      = (const float*)d_in[0];   // [1,1024,128]
  const float* adjC   = (const float*)d_in[1];   // [1024,1024]
  const float* adjE   = (const float*)d_in[2];   // [1024,1024]
  // d_in[3] edge_feat_comm: unused
  const float* Wself  = (const float*)d_in[4];   // [128,128]
  const float* Wcomm  = (const float*)d_in[5];
  const float* Wexec  = (const float*)d_in[6];
  const float* bias   = (const float*)d_in[7];   // [128]
  const float* Wa1    = (const float*)d_in[8];   // [256,128]
  const float* ba1    = (const float*)d_in[9];   // [128]
  const float* Wa2    = (const float*)d_in[10];  // [128]
  const float* ba2    = (const float*)d_in[11];  // [1]
  const float* gamma  = (const float*)d_in[12];  // [128]
  const float* beta   = (const float*)d_in[13];  // [128]
  float* out = (float*)d_out;

  const size_t ND = (size_t)NN * DD;     // 131072
  const size_t WW = (size_t)DD * DD;     // 16384
  const size_t AA = (size_t)NN * NN;     // 1048576

  char* base = (char*)d_ws;
  float* S0   = (float*)base;  base += ND * 4;
  float* SRC  = (float*)base;  base += ND * 4;
  float* TGT  = (float*)base;  base += ND * 4;
  float* PRE  = (float*)base;  base += ND * 4;
  float* MEAN = (float*)base;  base += DD * 4;
  float* RSTD = (float*)base;  base += DD * 4;
  __bf16* Xb     = (__bf16*)base; base += ND * 2;
  __bf16* Tb     = (__bf16*)base; base += ND * 2;
  __bf16* Eb     = (__bf16*)base; base += ND * 2;
  __bf16* Wselfb = (__bf16*)base; base += WW * 2;
  __bf16* Wcommb = (__bf16*)base; base += WW * 2;
  __bf16* Wexecb = (__bf16*)base; base += WW * 2;
  __bf16* Wsrcb  = (__bf16*)base; base += WW * 2;
  __bf16* Wtgtb  = (__bf16*)base; base += WW * 2;
  __bf16* ATTb   = (__bf16*)base; base += AA * 2;
  __bf16* ADJb   = (__bf16*)base; base += AA * 2;

  // K-1: one-time f32 -> bf16 conversions
  k_cvt4<<<(int)(ND / 4 / 256), 256, 0, stream>>>(X, Xb, (int)(ND / 4));
  k_cvt4<<<(int)(WW / 4 / 256), 256, 0, stream>>>(Wself, Wselfb, (int)(WW / 4));
  k_cvt4<<<(int)(WW / 4 / 256), 256, 0, stream>>>(Wcomm, Wcommb, (int)(WW / 4));
  k_cvt4<<<(int)(WW / 4 / 256), 256, 0, stream>>>(Wexec, Wexecb, (int)(WW / 4));
  k_cvt4<<<(int)(AA / 4 / 256), 256, 0, stream>>>(adjE, ADJb, (int)(AA / 4));

  // K0: compose Wsrc / Wtgt (2 * 128*128 outputs), bf16 store
  k_compose<<<128, 256, 0, stream>>>(Wcomm, Wa1, Wsrcb, Wtgtb);

  // K1: five [1024,128]x[128,128] WMMA GEMMs (2560 wave-tiles / 8 waves per block)
  k_gemm_x5<<<320, 256, 0, stream>>>(Xb, Wselfb, Wcommb, Wexecb, Wsrcb, Wtgtb,
                                     bias, ba1, S0, Tb, Eb, SRC, TGT);

  // K2: pairwise tanh attention + softmax (one block per row)
  k_att<<<NN, 256, 0, stream>>>(SRC, TGT, adjC, Wa2, ba2, ATTb);

  // K3: PRE = S0 + ATT@T + adjE@E (128 wave-strips / 8 waves per block)
  k_out_gemm<<<16, 256, 0, stream>>>(ATTb, Tb, ADJb, Eb, S0, PRE);

  // K4/K5: BatchNorm (training stats) + ReLU
  k_stats<<<1, DD, 0, stream>>>(PRE, MEAN, RSTD);
  k_bn_relu<<<(int)(ND / 256), 256, 0, stream>>>(PRE, MEAN, RSTD, gamma, beta, out);
}
